// RecurrentModel_74869869904559
// MI455X (gfx1250) — matrix operand order repaired and checked by reading, not compiled, not run on hardware
//
#include <hip/hip_runtime.h>

// ---------------- Types ----------------
typedef __bf16 bf16_t;
typedef __attribute__((ext_vector_type(16))) __bf16 v16bf;
typedef __attribute__((ext_vector_type(8)))  float  v8f;

// ---------------- Problem constants ----------------
static constexpr int Bn    = 2048;
static constexpr int DETER = 8192;
static constexpr int STOCH = 1024;
static constexpr int ACT   = 32;
static constexpr int ACTP  = 64;                 // zero-padded K for the action GEMM
static constexpr int HID   = 1024;
static constexpr int G     = 8;
static constexpr int DPG   = DETER / G;          // 1024
static constexpr int INBLK = 3 * HID + DPG;      // 4096

static constexpr int KT  = 64;                   // K elements staged per LDS stage
static constexpr int BTS = 72;                   // padded LDS row stride (bank-conflict free)

// ---------------- Scalar helpers ----------------
__device__ __forceinline__ bf16_t f2bf(float f) {
  union { float f; unsigned u; } in; in.f = f;
  unsigned r = (in.u + 0x7FFFu + ((in.u >> 16) & 1u)) >> 16;
  union { unsigned short s; bf16_t b; } out; out.s = (unsigned short)r;
  return out.b;
}
__device__ __forceinline__ float sigmoidf_(float x) { return 1.0f / (1.0f + __expf(-x)); }

__device__ __forceinline__ v8f wmma_bf16(v16bf a, v16bf b, v8f c) {
  return __builtin_amdgcn_wmma_f32_16x16x32_bf16(false, a, false, b, (short)0, c, false, false);
}

// A-fragment (16x32 bf16) direct from global, row-major [.., lda]:
// lane 0..15 : row m0+l,    K = k0+0..7  and k0+16..23
// lane 16..31: row m0+l-16, K = k0+8..15 and k0+24..31
__device__ __forceinline__ v16bf load_frag_a(const bf16_t* __restrict__ A, int lda,
                                             int m0, int k0, int lane) {
  int row = m0 + (lane & 15);
  int kb  = k0 + ((lane >> 4) << 3);
  union { uint4 u[2]; v16bf v; } f;
  f.u[0] = *(const uint4*)(A + (size_t)row * lda + kb);
  f.u[1] = *(const uint4*)(A + (size_t)row * lda + kb + 16);
  return f.v;
}

// B-fragment (32x16 bf16) from LDS tile lds[col][BTS]:
// lane l: column colBase + (l&15); K = klocal + (l<16?0:8) + 0..7 and +16..23
__device__ __forceinline__ v16bf frag_b_lds(const bf16_t* lds, int colBase, int klocal,
                                            int lane) {
  int col = colBase + (lane & 15);
  int kb  = klocal + ((lane >> 4) << 3);
  const uint4* p = (const uint4*)(lds + col * BTS + kb);
  union { uint4 u[2]; v16bf v; } f;
  f.u[0] = p[0];
  f.u[1] = p[2];
  return f.v;
}

// Async-DMA stage of a 64-column x KT tile of WT[N][K] into LDS (512 x 16B chunks,
// 2 per thread => ASYNCcnt +2 per wave per stage).
__device__ __forceinline__ void stage_b64_async(bf16_t* lds, const bf16_t* __restrict__ WT,
                                                int ldk, int n0, int k0) {
  #pragma unroll
  for (int i = 0; i < 2; ++i) {
    int c   = threadIdx.x + i * 256;       // 0..511
    int col = c >> 3;                      // 0..63
    int kch = (c & 7) * 8;                 // 0..56, 8 bf16 = 16B per chunk
    const bf16_t* g = WT + (size_t)(n0 + col) * ldk + k0 + kch;
    unsigned l = (unsigned)(size_t)(lds + col * BTS + kch);
    asm volatile("global_load_async_to_lds_b128 %0, %1, off"
                 :: "v"(l), "v"((unsigned long long)(size_t)g) : "memory");
  }
}

// Async-DMA stage of 3 x 64-column chunks (reset/cand/update) x KT into LDS
// (1536 chunks, 6 per thread => ASYNCcnt +6 per wave per stage).
__device__ __forceinline__ void stage_b192_async(bf16_t* lds, const bf16_t* __restrict__ WTg,
                                                 int n0, int k0) {
  #pragma unroll
  for (int i = 0; i < 6; ++i) {
    int c    = threadIdx.x + i * 256;      // 0..1535
    int q    = c >> 3;                     // LDS column 0..191
    int kch  = (c & 7) * 8;
    int gcol = (q >> 6) * DPG + n0 + (q & 63);
    const bf16_t* g = WTg + (size_t)gcol * DPG + k0 + kch;
    unsigned l = (unsigned)(size_t)(lds + q * BTS + kch);
    asm volatile("global_load_async_to_lds_b128 %0, %1, off"
                 :: "v"(l), "v"((unsigned long long)(size_t)g) : "memory");
  }
}

__device__ __forceinline__ float block_sum256(float v) {
  __shared__ float sbuf[8];
  #pragma unroll
  for (int o = 16; o > 0; o >>= 1) v += __shfl_down(v, o, 32);
  int lane = threadIdx.x & 31, w = threadIdx.x >> 5;
  if (lane == 0) sbuf[w] = v;
  __syncthreads();
  if (w == 0) {
    float t = (lane < 8) ? sbuf[lane] : 0.0f;
    #pragma unroll
    for (int o = 4; o > 0; o >>= 1) t += __shfl_down(t, o, 32);
    if (lane == 0) sbuf[0] = t;
  }
  __syncthreads();
  return sbuf[0];
}

// ---------------- Elementwise / conversion kernels ----------------
__global__ __launch_bounds__(256) void cast_bf16_k(const float* __restrict__ in,
                                                   bf16_t* __restrict__ out, long n) {
  long i = (long)blockIdx.x * blockDim.x + threadIdx.x;
  long stride = (long)gridDim.x * blockDim.x;
  for (; i < n; i += stride) out[i] = f2bf(in[i]);
}

// action normalize + zero-pad K: 32 -> 64 (out is [Bn][ACTP])
__global__ __launch_bounds__(256) void act_norm_k(const float* __restrict__ a,
                                                  bf16_t* __restrict__ out) {
  int i = blockIdx.x * blockDim.x + threadIdx.x;  // over Bn*ACTP
  int b = i >> 6, k = i & 63;
  float r = 0.0f;
  if (k < ACT) {
    float v = a[b * ACT + k];
    r = v / fmaxf(1.0f, fabsf(v));
  }
  out[i] = f2bf(r);
}

// in[g][K][N] (row-major fp32) -> out[g][N][ldkOut] bf16, zero-padding K..ldkOut-1
__global__ __launch_bounds__(256) void transpose_bf16_k(const float* __restrict__ in,
                                                        bf16_t* __restrict__ out,
                                                        int K, int N, int ldkOut) {
  size_t baseIn  = (size_t)blockIdx.z * K * N;
  size_t baseOut = (size_t)blockIdx.z * N * ldkOut;
  size_t total   = (size_t)N * ldkOut;
  for (size_t i = (size_t)blockIdx.x * blockDim.x + threadIdx.x; i < total;
       i += (size_t)gridDim.x * blockDim.x) {
    size_t n = i / ldkOut, k = i % ldkOut;
    out[baseOut + i] = (k < (size_t)K) ? f2bf(in[baseIn + k * N + n]) : f2bf(0.0f);
  }
}

// rmsnorm(y_seg + bias, gain) -> silu -> xc bf16. One block per (b, seg) row of 1024.
__global__ __launch_bounds__(256) void rmsnorm_silu_xc_k(
    const float* __restrict__ y, const float* __restrict__ b1, const float* __restrict__ b2,
    const float* __restrict__ b3, const float* __restrict__ g1, const float* __restrict__ g2,
    const float* __restrict__ g3, bf16_t* __restrict__ xc) {
  int rid = blockIdx.x;
  int b = rid / 3, seg = rid % 3;
  const float* bias = (seg == 0) ? b1 : (seg == 1) ? b2 : b3;
  const float* gain = (seg == 0) ? g1 : (seg == 1) ? g2 : g3;
  const float* row = y + (size_t)b * (3 * HID) + seg * HID;
  float vals[4], ss = 0.0f;
  #pragma unroll
  for (int i = 0; i < 4; ++i) {
    int t = threadIdx.x + i * 256;
    float v = row[t] + bias[t];
    vals[i] = v;
    ss += v * v;
  }
  float inv = rsqrtf(block_sum256(ss) * (1.0f / HID) + 1e-6f);
  #pragma unroll
  for (int i = 0; i < 4; ++i) {
    int t = threadIdx.x + i * 256;
    float xn = vals[i] * inv * gain[t];
    xc[(size_t)b * (3 * HID) + seg * HID + t] = f2bf(xn * sigmoidf_(xn));
  }
}

// rmsnorm over 8192 with gain -> bf16. One block per batch row.
__global__ __launch_bounds__(256) void rmsnorm_h_k(const float* __restrict__ ydyn,
                                                   const float* __restrict__ gdyn,
                                                   bf16_t* __restrict__ h) {
  int b = blockIdx.x;
  const float* row = ydyn + (size_t)b * DETER;
  float ss = 0.0f;
  #pragma unroll
  for (int i = 0; i < 32; ++i) {
    float v = row[threadIdx.x + i * 256];
    ss += v * v;
  }
  float inv = rsqrtf(block_sum256(ss) * (1.0f / DETER) + 1e-6f);
  #pragma unroll
  for (int i = 0; i < 32; ++i) {
    int t = threadIdx.x + i * 256;
    h[(size_t)b * DETER + t] = f2bf(row[t] * inv * gdyn[t]);
  }
}

// ---------------- WMMA GEMM kernels ----------------
// Generic: C[m, colOff+n] = A[M,K](bf16 row-major) x WT[N,K](bf16).
// WG tile 256x64 (8 waves x 32x64); B staged in LDS via async DMA, double buffered.
// B fragments for each k-step are batch-loaded so DS latency hides under the WMMA burst.
__global__ __launch_bounds__(256) void gemm_bf16_k(const bf16_t* __restrict__ A, int lda,
                                                   const bf16_t* __restrict__ WT, int K,
                                                   float* __restrict__ C, int ldc, int colOff) {
  __shared__ bf16_t bt[2][64 * BTS];
  const int lane  = threadIdx.x & 31;
  const int wave  = threadIdx.x >> 5;
  const int mBase = blockIdx.y * 256 + wave * 32;
  const int n0    = blockIdx.x * 64;
  v8f acc[2][4] = {};
  const int nstages = K / KT;
  stage_b64_async(&bt[0][0], WT, K, n0, 0);
  for (int s = 0; s < nstages; ++s) {
    const int k0 = s * KT;
    if (s + 1 < nstages) {
      stage_b64_async(&bt[(s + 1) & 1][0], WT, K, n0, k0 + KT);
      asm volatile("s_wait_asynccnt 0x2" ::: "memory");
    } else {
      asm volatile("s_wait_asynccnt 0x0" ::: "memory");
    }
    __syncthreads();
    #pragma unroll
    for (int kl = 0; kl < KT; kl += 32) {
      if (kl == 0 && s + 1 < nstages)
        __builtin_prefetch(A + (size_t)(mBase + (lane & 15)) * lda + k0 + KT, 0, 1);
      v16bf a0 = load_frag_a(A, lda, mBase, k0 + kl, lane);
      v16bf a1 = load_frag_a(A, lda, mBase + 16, k0 + kl, lane);
      v16bf bv[4];
      #pragma unroll
      for (int j = 0; j < 4; ++j) bv[j] = frag_b_lds(&bt[s & 1][0], j * 16, kl, lane);
      #pragma unroll
      for (int j = 0; j < 4; ++j) {
        acc[0][j] = wmma_bf16(a0, bv[j], acc[0][j]);
        acc[1][j] = wmma_bf16(a1, bv[j], acc[1][j]);
      }
    }
    __syncthreads();
  }
  const int mOff = (lane >> 4) << 3, nl = lane & 15;
  #pragma unroll
  for (int i = 0; i < 2; ++i)
    #pragma unroll
    for (int j = 0; j < 4; ++j)
      #pragma unroll
      for (int r = 0; r < 8; ++r)
        C[(size_t)(mBase + i * 16 + mOff + r) * ldc + colOff + n0 + j * 16 + nl] = acc[i][j][r];
}

// Block-diagonal dyn GEMM: A gathered on the fly from xc (k<3072) / deter slice (k>=3072).
__global__ __launch_bounds__(256) void gemm_dyn_k(const bf16_t* __restrict__ xc,
                                                  const bf16_t* __restrict__ deter_bf,
                                                  const bf16_t* __restrict__ WdynT,
                                                  float* __restrict__ ydyn) {
  __shared__ bf16_t bt[2][64 * BTS];
  const int lane  = threadIdx.x & 31;
  const int wave  = threadIdx.x >> 5;
  const int mBase = blockIdx.y * 256 + wave * 32;
  const int n0    = blockIdx.x * 64;
  const int g     = blockIdx.z;
  const bf16_t* __restrict__ WTg = WdynT + (size_t)g * DPG * INBLK;
  v8f acc[2][4] = {};
  const int nstages = INBLK / KT;
  stage_b64_async(&bt[0][0], WTg, INBLK, n0, 0);
  for (int s = 0; s < nstages; ++s) {
    const int k0 = s * KT;
    if (s + 1 < nstages) {
      stage_b64_async(&bt[(s + 1) & 1][0], WTg, INBLK, n0, k0 + KT);
      asm volatile("s_wait_asynccnt 0x2" ::: "memory");
    } else {
      asm volatile("s_wait_asynccnt 0x0" ::: "memory");
    }
    __syncthreads();
    // region select: stage boundaries (mult of 64) never straddle the 3072 split
    const bf16_t* Ab;
    int lda, kk;
    if (k0 < 3 * HID) { Ab = xc; lda = 3 * HID; kk = k0; }
    else              { Ab = deter_bf + g * DPG; lda = DETER; kk = k0 - 3 * HID; }
    #pragma unroll
    for (int kl = 0; kl < KT; kl += 32) {
      v16bf a0 = load_frag_a(Ab, lda, mBase, kk + kl, lane);
      v16bf a1 = load_frag_a(Ab, lda, mBase + 16, kk + kl, lane);
      v16bf bv[4];
      #pragma unroll
      for (int j = 0; j < 4; ++j) bv[j] = frag_b_lds(&bt[s & 1][0], j * 16, kl, lane);
      #pragma unroll
      for (int j = 0; j < 4; ++j) {
        acc[0][j] = wmma_bf16(a0, bv[j], acc[0][j]);
        acc[1][j] = wmma_bf16(a1, bv[j], acc[1][j]);
      }
    }
    __syncthreads();
  }
  const int mOff = (lane >> 4) << 3, nl = lane & 15;
  #pragma unroll
  for (int i = 0; i < 2; ++i)
    #pragma unroll
    for (int j = 0; j < 4; ++j)
      #pragma unroll
      for (int r = 0; r < 8; ++r)
        ydyn[(size_t)(mBase + i * 16 + mOff + r) * DETER + g * DPG + n0 + j * 16 + nl] =
            acc[i][j][r];
}

// Fused out-GEMM + GRU gating: wave tile 16x64 with 3 accumulator sets
// (reset/cand/update columns), gating applied in the epilogue -> next_deter.
__global__ __launch_bounds__(256) void gemm_out_gate_k(const bf16_t* __restrict__ h,
                                                       const bf16_t* __restrict__ WoutT,
                                                       const float* __restrict__ deter,
                                                       float* __restrict__ out) {
  __shared__ bf16_t bt[2][192 * BTS];
  const int lane  = threadIdx.x & 31;
  const int wave  = threadIdx.x >> 5;
  const int mBase = blockIdx.y * 128 + wave * 16;
  const int n0    = blockIdx.x * 64;
  const int g     = blockIdx.z;
  const bf16_t* __restrict__ WTg = WoutT + (size_t)g * (3 * DPG) * DPG;
  const bf16_t* __restrict__ Ab  = h + g * DPG;  // row stride DETER
  v8f acc[3][4] = {};
  const int nstages = DPG / KT;
  stage_b192_async(&bt[0][0], WTg, n0, 0);
  for (int s = 0; s < nstages; ++s) {
    const int k0 = s * KT;
    if (s + 1 < nstages) {
      stage_b192_async(&bt[(s + 1) & 1][0], WTg, n0, k0 + KT);
      asm volatile("s_wait_asynccnt 0x6" ::: "memory");
    } else {
      asm volatile("s_wait_asynccnt 0x0" ::: "memory");
    }
    __syncthreads();
    #pragma unroll
    for (int kl = 0; kl < KT; kl += 32) {
      v16bf a0 = load_frag_a(Ab, DETER, mBase, k0 + kl, lane);
      #pragma unroll
      for (int c = 0; c < 3; ++c) {
        v16bf bv[4];
        #pragma unroll
        for (int j = 0; j < 4; ++j)
          bv[j] = frag_b_lds(&bt[s & 1][0], c * 64 + j * 16, kl, lane);
        #pragma unroll
        for (int j = 0; j < 4; ++j) acc[c][j] = wmma_bf16(a0, bv[j], acc[c][j]);
      }
    }
    __syncthreads();
  }
  const int mOff = (lane >> 4) << 3, nl = lane & 15;
  #pragma unroll
  for (int j = 0; j < 4; ++j)
    #pragma unroll
    for (int r = 0; r < 8; ++r) {
      int m = mBase + mOff + r;
      int n = n0 + j * 16 + nl;
      float reset = sigmoidf_(acc[0][j][r]);
      float cand  = tanhf(reset * acc[1][j][r]);
      float upd   = sigmoidf_(acc[2][j][r] - 1.0f);
      size_t idx  = (size_t)m * DETER + g * DPG + n;
      out[idx] = upd * cand + (1.0f - upd) * deter[idx];
    }
}

// ---------------- Host launcher ----------------
extern "C" void kernel_launch(void* const* d_in, const int* in_sizes, int n_in,
                              void* d_out, int out_size, void* d_ws, size_t ws_size,
                              hipStream_t stream) {
  const float* deter  = (const float*)d_in[0];
  const float* stoch  = (const float*)d_in[1];
  const float* action = (const float*)d_in[2];
  const float* W1 = (const float*)d_in[3];
  const float* b1 = (const float*)d_in[4];
  const float* W2 = (const float*)d_in[5];
  const float* b2 = (const float*)d_in[6];
  const float* W3 = (const float*)d_in[7];
  const float* b3 = (const float*)d_in[8];
  const float* g1 = (const float*)d_in[9];
  const float* g2 = (const float*)d_in[10];
  const float* g3 = (const float*)d_in[11];
  const float* Wdyn = (const float*)d_in[12];
  const float* gdyn = (const float*)d_in[13];
  const float* Wout = (const float*)d_in[14];
  float* out = (float*)d_out;

  char* ws = (char*)d_ws;
  size_t off = 0;
  auto carve = [&](size_t bytes) -> void* {
    void* p = ws + off;
    off += (bytes + 255) & ~(size_t)255;
    return p;
  };
  bf16_t* deter_bf = (bf16_t*)carve((size_t)Bn * DETER * 2);
  bf16_t* stoch_bf = (bf16_t*)carve((size_t)Bn * STOCH * 2);
  bf16_t* a_bf     = (bf16_t*)carve((size_t)Bn * ACTP * 2);
  bf16_t* W1T      = (bf16_t*)carve((size_t)HID * DETER * 2);
  bf16_t* W2T      = (bf16_t*)carve((size_t)HID * STOCH * 2);
  bf16_t* W3T      = (bf16_t*)carve((size_t)HID * ACTP * 2);
  bf16_t* WdynT    = (bf16_t*)carve((size_t)G * DPG * INBLK * 2);
  bf16_t* WoutT    = (bf16_t*)carve((size_t)G * 3 * DPG * DPG * 2);
  float*  y123     = (float*)carve((size_t)Bn * 3 * HID * 4);
  bf16_t* xc       = (bf16_t*)carve((size_t)Bn * 3 * HID * 2);
  float*  ydyn     = (float*)carve((size_t)Bn * DETER * 4);
  bf16_t* h_bf     = (bf16_t*)carve((size_t)Bn * DETER * 2);
  (void)ws_size; (void)in_sizes; (void)n_in; (void)out_size;

  // 1) precision conversion (per call, stateless)
  cast_bf16_k<<<4096, 256, 0, stream>>>(deter, deter_bf, (long)Bn * DETER);
  cast_bf16_k<<<1024, 256, 0, stream>>>(stoch, stoch_bf, (long)Bn * STOCH);
  act_norm_k<<<(Bn * ACTP) / 256, 256, 0, stream>>>(action, a_bf);
  transpose_bf16_k<<<dim3(2048, 1, 1), 256, 0, stream>>>(W1, W1T, DETER, HID, DETER);
  transpose_bf16_k<<<dim3(512, 1, 1), 256, 0, stream>>>(W2, W2T, STOCH, HID, STOCH);
  transpose_bf16_k<<<dim3(32, 1, 1), 256, 0, stream>>>(W3, W3T, ACT, HID, ACTP);
  transpose_bf16_k<<<dim3(1024, 1, G), 256, 0, stream>>>(Wdyn, WdynT, INBLK, DPG, INBLK);
  transpose_bf16_k<<<dim3(1024, 1, G), 256, 0, stream>>>(Wout, WoutT, DPG, 3 * DPG, DPG);

  // 2) input projections -> y123 (B, 3*HID) raw
  gemm_bf16_k<<<dim3(HID / 64, Bn / 256), 256, 0, stream>>>(deter_bf, DETER, W1T, DETER,
                                                            y123, 3 * HID, 0);
  gemm_bf16_k<<<dim3(HID / 64, Bn / 256), 256, 0, stream>>>(stoch_bf, STOCH, W2T, STOCH,
                                                            y123, 3 * HID, HID);
  gemm_bf16_k<<<dim3(HID / 64, Bn / 256), 256, 0, stream>>>(a_bf, ACTP, W3T, ACTP,
                                                            y123, 3 * HID, 2 * HID);

  // 3) bias + rmsnorm + silu -> xc bf16
  rmsnorm_silu_xc_k<<<Bn * 3, 256, 0, stream>>>(y123, b1, b2, b3, g1, g2, g3, xc);

  // 4) block-diagonal dyn GEMM (A gathered from xc + deter slice)
  gemm_dyn_k<<<dim3(DPG / 64, Bn / 256, G), 256, 0, stream>>>(xc, deter_bf, WdynT, ydyn);

  // 5) rmsnorm(gdyn) -> h bf16
  rmsnorm_h_k<<<Bn, 256, 0, stream>>>(ydyn, gdyn, h_bf);

  // 6) block-diagonal out GEMM fused with GRU gating -> next_deter
  gemm_out_gate_k<<<dim3(DPG / 64, Bn / 128, G), 256, 0, stream>>>(h_bf, WoutT, deter, out);
}
